// FHEBlock_7679401525931
// MI455X (gfx1250) — matrix operand order repaired and checked by reading, not compiled, not run on hardware
//
#include <hip/hip_runtime.h>

// ---------------------------------------------------------------------------
// CDNA5 (gfx1250) FHE transformer block: all GEMMs via v_wmma_f32_16x16x32_bf16,
// LDS double-buffered with async global->LDS loads (ASYNCcnt) when available.
// Weights are converted f32->bf16 and transposed ONCE so every GEMM uses the
// same [N,K] direct-copy staging path (no per-tile transpose in the hot loop).
// ---------------------------------------------------------------------------

typedef __attribute__((ext_vector_type(16))) __bf16 v16bf;
typedef __attribute__((ext_vector_type(8)))  float  v8f;

#if __has_builtin(__builtin_amdgcn_global_load_async_to_lds_b128)
#define HAS_ASYNC 1
#else
#define HAS_ASYNC 0
#endif

#if __has_builtin(__builtin_amdgcn_s_wait_asynccnt)
#define WAIT_ASYNC(n) __builtin_amdgcn_s_wait_asynccnt(n)
#else
#define WAIT_ASYNC(n) asm volatile("s_wait_asynccnt %0" ::"i"(n) : "memory")
#endif

// builtin expects native int __vector(4) pointers in global (AS1) / LDS (AS3)
typedef int nat_int4 __attribute__((vector_size(16)));
typedef __attribute__((address_space(1))) nat_int4 gas_int4;
typedef __attribute__((address_space(3))) nat_int4 las_int4;

#define BM 128
#define BN 128
#define BK 64
#define LDT 72        // padded LDS row stride (elems): 144B, 16B-aligned, conflict-free
#define PER_TILE 8    // async b128 ops issued per thread per staged tile (4 A + 4 B)

union Frag { v16bf v; int4 q[2]; };

// Stage one BM x BK (A) and BN x BK (B) bf16 tile into LDS ([row][k], padded).
__device__ __forceinline__ void stage_tile(const __bf16* __restrict__ gA, int lda,
                                           const __bf16* __restrict__ gB, int ldb,
                                           __bf16* lsA, __bf16* lsB, int t)
{
#pragma unroll
    for (int j = 0; j < 4; ++j) {
        const int c   = t + j * 256;      // 1024 16B-chunks per tile
        const int row = c >> 3;           // 8 chunks per 64-elem row
        const int o16 = (c & 7) * 8;      // elem offset within row
        const __bf16* sa = gA + (size_t)row * lda + o16;
        const __bf16* sb = gB + (size_t)row * ldb + o16;
        __bf16* da = lsA + row * LDT + o16;
        __bf16* db = lsB + row * LDT + o16;
#if HAS_ASYNC
        __builtin_amdgcn_global_load_async_to_lds_b128((gas_int4*)sa, (las_int4*)da, 0, 0);
        __builtin_amdgcn_global_load_async_to_lds_b128((gas_int4*)sb, (las_int4*)db, 0, 0);
#else
        *(int4*)da = *(const int4*)sa;
        *(int4*)db = *(const int4*)sb;
#endif
    }
}

// C = act(scale * (A @ B^T)) [+ Res]
//  A: bf16 [M,K] row-major; B: bf16 [N,K] row-major (i.e. B^T of the math GEMM)
//  ACT: 0 = none, 1 = y*y + y.  OUTTRANS: write C[col*M + row] (for v^T).
template<int ACT, bool RES, bool OUTBF, bool OUTTRANS>
__global__ __launch_bounds__(256) void gemm_wmma(
    const __bf16* __restrict__ A, const __bf16* __restrict__ B,
    void* __restrict__ Cp, const float* __restrict__ Res,
    int M, int N, int K, float scale,
    long long sA, long long sB, long long sC, long long sRes)
{
    __shared__ __bf16 lsA[2][BM * LDT];
    __shared__ __bf16 lsB[2][BN * LDT];

    const int z = blockIdx.z;
    A += (size_t)z * sA;
    B += (size_t)z * sB;
    float*  Cf = (float*)Cp  + (size_t)z * sC;
    __bf16* Cb = (__bf16*)Cp + (size_t)z * sC;
    if (RES) Res += (size_t)z * sRes;

    const int m0 = blockIdx.y * BM;
    const int n0 = blockIdx.x * BN;
    const int t    = threadIdx.x;
    const int lane = t & 31;
    const int half = lane >> 4;        // which 16-lane half of the wave
    const int lrow = lane & 15;
    const int wid  = t >> 5;           // wave 0..7
    const int wm   = (wid & 3) * 32;   // wave row offset (4 waves in M)
    const int wn   = (wid >> 2) * 64;  // wave col offset (2 waves in N)

    const __bf16* gA = A + (size_t)m0 * K;
    const __bf16* gB = B + (size_t)n0 * K;

    v8f acc[2][4] = {};

    const int ktiles = K / BK;
    stage_tile(gA, K, gB, K, lsA[0], lsB[0], t);   // prologue: tile 0

    for (int kt = 0; kt < ktiles; ++kt) {
        const int cur = kt & 1;
        if (kt + 1 < ktiles) {
            stage_tile(gA + (kt + 1) * BK, K, gB + (kt + 1) * BK, K,
                       lsA[cur ^ 1], lsB[cur ^ 1], t);
#if HAS_ASYNC
            WAIT_ASYNC(PER_TILE);   // tile kt complete; kt+1 still in flight
#endif
        } else {
#if HAS_ASYNC
            WAIT_ASYNC(0);
#endif
        }
        __syncthreads();

        const __bf16* la = lsA[cur];
        const __bf16* lb = lsB[cur];
#pragma unroll
        for (int ks = 0; ks < 2; ++ks) {           // two 32-deep WMMA steps
            const int kb = ks * 32;
            Frag fa[2], fb[4];
#pragma unroll
            for (int i = 0; i < 2; ++i) {
                // A 16x32: lanes 0-15 K{0..7,16..23}, lanes 16-31 K{8..15,24..31}
                const __bf16* p = la + (wm + i * 16 + lrow) * LDT + kb;
                fa[i].q[0] = *(const int4*)(p + half * 8);
                fa[i].q[1] = *(const int4*)(p + 16 + half * 8);
            }
#pragma unroll
            for (int j = 0; j < 4; ++j) {
                // B 32x16: lane = column; lanes 0-15 K{0..7,16..23}, lanes 16-31 K{8..15,24..31}
                const __bf16* p = lb + (wn + j * 16 + lrow) * LDT + kb;
                fb[j].q[0] = *(const int4*)(p + half * 8);
                fb[j].q[1] = *(const int4*)(p + 16 + half * 8);
            }
#pragma unroll
            for (int i = 0; i < 2; ++i)
#pragma unroll
                for (int j = 0; j < 4; ++j)
                    acc[i][j] = __builtin_amdgcn_wmma_f32_16x16x32_bf16(
                        false, fa[i].v, false, fb[j].v, (short)0, acc[i][j], false, false);
        }
        __syncthreads();
    }

    // epilogue: C/D layout -> VGPR r: row = 8*half + r, col = lrow
#pragma unroll
    for (int i = 0; i < 2; ++i)
#pragma unroll
        for (int j = 0; j < 4; ++j)
#pragma unroll
            for (int r = 0; r < 8; ++r) {
                const int grow = m0 + wm + i * 16 + half * 8 + r;
                const int gcol = n0 + wn + j * 16 + lrow;
                const size_t idx = OUTTRANS ? ((size_t)gcol * M + grow)
                                            : ((size_t)grow * N + gcol);
                float y = acc[i][j][r] * scale;
                if (ACT == 1) y = y * y + y;
                if (RES) y += Res[idx];
                if (OUTBF) Cb[idx] = (__bf16)y;
                else       Cf[idx] = y;
            }
}

// Wt[n*K + k] = bf16(W[k*N + n])  -- one-time weight convert+transpose
__global__ __launch_bounds__(256) void transpose_w_kernel(
    const float* __restrict__ W, __bf16* __restrict__ Wt, int K, int N)
{
    __shared__ float tile[32][33];
    const int bn = blockIdx.x * 32;
    const int bk = blockIdx.y * 32;
    const int tx = threadIdx.x & 31, ty = threadIdx.x >> 5;
#pragma unroll
    for (int i = ty; i < 32; i += 8)
        tile[i][tx] = W[(size_t)(bk + i) * N + bn + tx];
    __syncthreads();
#pragma unroll
    for (int i = ty; i < 32; i += 8)
        Wt[(size_t)(bn + i) * K + bk + tx] = (__bf16)tile[tx][i];
}

// out_bf16[i] = bf16(x[i] * g[i % D])
__global__ __launch_bounds__(256) void scale_bf16_kernel(
    const float* __restrict__ x, const float* __restrict__ g,
    __bf16* __restrict__ out, size_t total, int D)
{
    size_t i = ((size_t)blockIdx.x * blockDim.x + threadIdx.x) * 4;
    if (i >= total) return;
    float4 xv = *(const float4*)(x + i);
    int d = (int)(i % (size_t)D);
    float4 gv = *(const float4*)(g + d);
    union { __bf16 h[4]; int2 v; } u;
    u.h[0] = (__bf16)(xv.x * gv.x);
    u.h[1] = (__bf16)(xv.y * gv.y);
    u.h[2] = (__bf16)(xv.z * gv.z);
    u.h[3] = (__bf16)(xv.w * gv.w);
    *(int2*)(out + i) = u.v;
}

extern "C" void kernel_launch(void* const* d_in, const int* in_sizes, int n_in,
                              void* d_out, int out_size, void* d_ws, size_t ws_size,
                              hipStream_t stream)
{
    (void)n_in; (void)out_size; (void)ws_size;
    const float* x  = (const float*)d_in[0];
    const float* Wq = (const float*)d_in[1];
    const float* Wk = (const float*)d_in[2];
    const float* Wv = (const float*)d_in[3];
    const float* Wo = (const float*)d_in[4];
    const float* W1 = (const float*)d_in[5];
    const float* W2 = (const float*)d_in[6];
    const float* g1 = (const float*)d_in[7];
    const float* g2 = (const float*)d_in[8];

    const int    D     = in_sizes[7];               // 1024
    const size_t total = (size_t)in_sizes[0];       // B*S*D
    const int    M     = (int)(total / (size_t)D);  // B*S = 8192
    const int    S     = 2048;
    const int    Bn    = M / S;                     // 4

    // workspace layout (reuse where lifetimes allow)
    char* ws = (char*)d_ws;
    const size_t szMD = (size_t)M * D * 2;             // bf16 [M,D]   16 MB
    const size_t szSS = (size_t)Bn * S * S * 2;        // bf16 [B,S,S] 33.5 MB
    __bf16* H   = (__bf16*)(ws);                       // h=x*g1 (later h2in)
    __bf16* Q   = (__bf16*)(ws + szMD);                // q      (later o)
    __bf16* Kb  = (__bf16*)(ws + 2 * szMD);            // k
    __bf16* Vt  = (__bf16*)(ws + 3 * szMD);            // v^T [B,D,S]
    __bf16* Ab  = (__bf16*)(ws + 4 * szMD);            // a [B,S,S] (later h2)
    float*  X1  = (float*)(ws + 4 * szMD + szSS);      // f32 residual x1
    char*   wt  = ws + 6 * szMD + szSS;                // transposed bf16 weights
    __bf16* WqT = (__bf16*)(wt);
    __bf16* WkT = (__bf16*)(wt + 2u * 1024 * 1024);
    __bf16* WvT = (__bf16*)(wt + 4u * 1024 * 1024);
    __bf16* WoT = (__bf16*)(wt + 6u * 1024 * 1024);
    __bf16* W1T = (__bf16*)(wt + 8u * 1024 * 1024);
    __bf16* W2T = (__bf16*)(wt + 12u * 1024 * 1024);
    __bf16* O    = Q;
    __bf16* H2in = H;
    __bf16* H2   = Ab;

    dim3 blk(256);
    const int n4 = (int)(total / 4);
    const long long SD = (long long)S * D, SS = (long long)S * S;

    // one-time weight convert + transpose: W[K,N] f32 -> Wt[N,K] bf16
    transpose_w_kernel<<<dim3(D / 32, D / 32), blk, 0, stream>>>(Wq, WqT, D, D);
    transpose_w_kernel<<<dim3(D / 32, D / 32), blk, 0, stream>>>(Wk, WkT, D, D);
    transpose_w_kernel<<<dim3(D / 32, D / 32), blk, 0, stream>>>(Wv, WvT, D, D);
    transpose_w_kernel<<<dim3(D / 32, D / 32), blk, 0, stream>>>(Wo, WoT, D, D);
    transpose_w_kernel<<<dim3(2 * D / 32, D / 32), blk, 0, stream>>>(W1, W1T, D, 2 * D);
    transpose_w_kernel<<<dim3(D / 32, 2 * D / 32), blk, 0, stream>>>(W2, W2T, 2 * D, D);

    // h = bf16(x * g1)
    scale_bf16_kernel<<<(n4 + 255) / 256, blk, 0, stream>>>(x, g1, H, total, D);
    // q = 0.1*(h@Wq), k = 0.1*(h@Wk)
    gemm_wmma<0, false, true, false><<<dim3(D / BN, M / BM, 1), blk, 0, stream>>>(
        H, WqT, Q, nullptr, M, D, D, 0.1f, 0, 0, 0, 0);
    gemm_wmma<0, false, true, false><<<dim3(D / BN, M / BM, 1), blk, 0, stream>>>(
        H, WkT, Kb, nullptr, M, D, D, 0.1f, 0, 0, 0, 0);
    // v^T[b][d][t] = (h@Wv)[b][t][d]  (transposed epilogue, per batch)
    gemm_wmma<0, false, true, true><<<dim3(D / BN, S / BM, Bn), blk, 0, stream>>>(
        H, WvT, Vt, nullptr, S, D, D, 1.0f, SD, 0, SD, 0);
    // a = poly(0.1 * q @ k^T)  per batch
    gemm_wmma<1, false, true, false><<<dim3(S / BN, S / BM, Bn), blk, 0, stream>>>(
        Q, Kb, Ab, nullptr, S, S, D, 0.1f, SD, SD, SS, 0);
    // o = a @ v  (= a @ (v^T)^T)  per batch; writes contiguous => flat [M,D]
    gemm_wmma<0, false, true, false><<<dim3(D / BN, S / BM, Bn), blk, 0, stream>>>(
        Ab, Vt, O, nullptr, S, D, S, 1.0f, SS, SD, SD, 0);
    // x1 = x + 0.1 * (o @ Wo)  (f32 out)
    gemm_wmma<0, true, false, false><<<dim3(D / BN, M / BM, 1), blk, 0, stream>>>(
        O, WoT, X1, x, M, D, D, 0.1f, 0, 0, 0, 0);
    // h2in = bf16(x1 * g2)
    scale_bf16_kernel<<<(n4 + 255) / 256, blk, 0, stream>>>(X1, g2, H2in, total, D);
    // h2 = poly(h2in @ W1)
    gemm_wmma<1, false, true, false><<<dim3(2 * D / BN, M / BM, 1), blk, 0, stream>>>(
        H2in, W1T, H2, nullptr, M, 2 * D, D, 1.0f, 0, 0, 0, 0);
    // out = x1 + h2 @ W2  (f32 out)
    gemm_wmma<0, true, false, false><<<dim3(D / BN, M / BM, 1), blk, 0, stream>>>(
        H2, W2T, (float*)d_out, X1, M, D, 2 * D, 1.0f, 0, 0, 0, 0);
}
